// HEARTForSequenceClassification_65146063946133
// MI455X (gfx1250) — compile-verified, matrix operand order checked by reading, not run on hardware
//
#include <hip/hip_runtime.h>

// ---------------- types ----------------
typedef unsigned short u16;
typedef __attribute__((ext_vector_type(16))) u16      v16u;
typedef __attribute__((ext_vector_type(16))) __bf16   v16bf;
typedef __attribute__((ext_vector_type(8)))  float    v8f;

union BFfrag { v16u u; v16bf b; };

struct __align__(16) US8 { u16 s[8]; };

// dims
#define BB 2
#define SS 384
#define DD 768
#define HH 12
#define EE 64
#define LL 6
#define FF 2048
#define DKK 64
#define BSS (BB*SS)

__device__ __forceinline__ u16 f2bf(float f) {
    unsigned u = __float_as_uint(f);
    unsigned r = (u + 0x7FFFu + ((u >> 16) & 1u)) >> 16;
    return (u16)r;
}

__device__ __forceinline__ float gelu_f(float x) {
    float x3 = x * x * x;
    return 0.5f * x * (1.0f + tanhf(0.7978845608028654f * (x + 0.044715f * x3)));
}

__device__ __forceinline__ void pack8(BFfrag& f, int base, float4 a, float4 b) {
    f.u[base+0] = f2bf(a.x); f.u[base+1] = f2bf(a.y);
    f.u[base+2] = f2bf(a.z); f.u[base+3] = f2bf(a.w);
    f.u[base+4] = f2bf(b.x); f.u[base+5] = f2bf(b.y);
    f.u[base+6] = f2bf(b.z); f.u[base+7] = f2bf(b.w);
}

// ---------------- generic bf16 WMMA GEMM ----------------
// C[m,n] = act( alpha * sum_k A[m,k]*Bt[n,k] + bias[n] + addsrc[m,n] )
// A fp32 row-major (lda), Bt bf16 row-major [N,K] (ldb), C fp32 (ldc).
// Block: 256 thr = 8 waves; block tile 128(M) x 64(N); K%32==0, M%128==0, N%64==0.
// per-z offsets: off = (z/zdiv)*s1 + (z%zdiv)*s2
__global__ void wmma_gemm_kernel(
    const float* __restrict__ A, long long az1, long long az2, int lda,
    const u16*  __restrict__ Bt, long long bz1, long long bz2, int ldb,
    float* __restrict__ C, long long cz1, long long cz2, int ldc,
    const float* __restrict__ addsrc, long long dz1, long long dz2, int ldd,
    const float* __restrict__ bias,
    float alpha, int act_gelu, int K, int zdiv)
{
    int z = blockIdx.z;
    long long zq = z / zdiv, zr = z % zdiv;
    const float* Az = A + zq * az1 + zr * az2;
    const u16*   Bz = Bt + zq * bz1 + zr * bz2;
    float*       Cz = C + zq * cz1 + zr * cz2;
    const float* Dz = addsrc ? (addsrc + zq * dz1 + zr * dz2) : (const float*)0;

    int tid = threadIdx.x;
    int wv = tid >> 5, lane = tid & 31;
    int g = lane >> 4, r = lane & 15;
    int m0 = blockIdx.y * 128 + wv * 16;
    int n0 = blockIdx.x * 64;

    v8f acc[4];
    #pragma unroll
    for (int j = 0; j < 4; j++) acc[j] = v8f{};

    const float* arow = Az + (long long)(m0 + r) * lda;
    const u16*   brow = Bz + (long long)(n0 + r) * ldb;

    for (int k = 0; k < K; k += 32) {
        int kb0 = k + 8 * g;
        int kb1 = kb0 + 16;
        BFfrag a;
        {
            float4 x0 = *(const float4*)(arow + kb0);
            float4 x1 = *(const float4*)(arow + kb0 + 4);
            float4 x2 = *(const float4*)(arow + kb1);
            float4 x3 = *(const float4*)(arow + kb1 + 4);
            pack8(a, 0, x0, x1);
            pack8(a, 8, x2, x3);
        }
        #pragma unroll
        for (int j = 0; j < 4; j++) {
            const u16* br = brow + (long long)(j * 16) * ldb;
            US8 u0 = *(const US8*)(br + kb0);
            US8 u1 = *(const US8*)(br + kb1);
            BFfrag bb;
            #pragma unroll
            for (int i = 0; i < 8; i++) { bb.u[i] = u0.s[i]; bb.u[8 + i] = u1.s[i]; }
            acc[j] = __builtin_amdgcn_wmma_f32_16x16x32_bf16(
                false, a.b, false, bb.b, (short)0, acc[j], false, false);
        }
    }

    #pragma unroll
    for (int j = 0; j < 4; j++) {
        int n = n0 + j * 16 + r;
        float bv = bias ? bias[n] : 0.0f;
        #pragma unroll
        for (int v = 0; v < 8; v++) {
            int m = m0 + v + 8 * g;
            float val = acc[j][v] * alpha + bv;
            if (Dz) val += Dz[(long long)m * ldd + n];
            if (act_gelu) val = gelu_f(val);
            Cz[(long long)m * ldc + n] = val;
        }
    }
}

// ---------------- weight convert / transpose ----------------
// dst[n*K + k] = bf16(src[k*N + n])  (src is [K,N] row-major)
__global__ void cvtT_kernel(const float* __restrict__ src, u16* __restrict__ dst,
                            int K, int N) {
    long long i = (long long)blockIdx.x * 256 + threadIdx.x;
    if (i >= (long long)K * N) return;
    long long n = i / K, k = i % K;
    dst[i] = f2bf(src[k * N + n]);
}

__global__ void cvtN_kernel(const float* __restrict__ s, u16* __restrict__ d, long long n) {
    long long i = (long long)blockIdx.x * 256 + threadIdx.x;
    if (i < n) d[i] = f2bf(s[i]);
}

// v [B,S,D] fp32 -> vt [B,H,DK,S] bf16
__global__ void transpose_v_kernel(const float* __restrict__ v, u16* __restrict__ vt) {
    long long i = (long long)blockIdx.x * 256 + threadIdx.x;
    if (i >= (long long)BB * HH * DKK * SS) return;
    long long z = i / (DKK * SS);
    long long rr = i % (DKK * SS);
    int dk = (int)(rr / SS), s = (int)(rr % SS);
    int b = (int)(z / HH), h = (int)(z % HH);
    vt[i] = f2bf(v[((long long)(b * SS + s)) * DD + h * DKK + dk]);
}

// ---------------- layernorm over D ----------------
__global__ void ln_rows_kernel(const float* __restrict__ x, const float* __restrict__ gg,
                               const float* __restrict__ bb, float* __restrict__ y, int Dn) {
    __shared__ float s1[256], s2[256];
    long long base = (long long)blockIdx.x * Dn;
    int tid = threadIdx.x;
    float a = 0.f, b = 0.f;
    for (int i = tid; i < Dn; i += 256) { float v = x[base + i]; a += v; b += v * v; }
    s1[tid] = a; s2[tid] = b; __syncthreads();
    for (int o = 128; o > 0; o >>= 1) {
        if (tid < o) { s1[tid] += s1[tid + o]; s2[tid] += s2[tid + o]; }
        __syncthreads();
    }
    float mean = s1[0] / Dn;
    float var = s2[0] / Dn - mean * mean;
    float rstd = rsqrtf(var + 1e-5f);
    for (int i = tid; i < Dn; i += 256)
        y[base + i] = (x[base + i] - mean) * rstd * gg[i] + bb[i];
}

// ---------------- softmax with mask (in-place) ----------------
__global__ void softmax_kernel(float* __restrict__ p, const unsigned char* __restrict__ mask) {
    __shared__ float red[256];
    long long row = blockIdx.x;             // b*H*S + h*S + n
    int b = (int)(row / (HH * SS));
    int n = (int)(row % SS);
    float* pr = p + row * SS;
    const unsigned char* mr = mask + ((long long)(b * SS) + n) * SS;
    int tid = threadIdx.x;
    float mx = -3.4e38f;
    for (int m = tid; m < SS; m += 256) {
        float v = mr[m] ? -3.4e38f : pr[m];
        mx = fmaxf(mx, v);
    }
    red[tid] = mx; __syncthreads();
    for (int o = 128; o > 0; o >>= 1) { if (tid < o) red[tid] = fmaxf(red[tid], red[tid + o]); __syncthreads(); }
    mx = red[0]; __syncthreads();
    float sm = 0.f;
    for (int m = tid; m < SS; m += 256) {
        float e = mr[m] ? 0.f : __expf(pr[m] - mx);
        pr[m] = e; sm += e;
    }
    red[tid] = sm; __syncthreads();
    for (int o = 128; o > 0; o >>= 1) { if (tid < o) red[tid] += red[tid + o]; __syncthreads(); }
    float inv = red[0] > 0.f ? 1.f / red[0] : 0.f;
    for (int m = tid; m < SS; m += 256) pr[m] *= inv;
}

// ---------------- per-token type transform (left/right) ----------------
__global__ void token_transform_kernel(const float* __restrict__ emb, const int* __restrict__ types,
                                       const float* __restrict__ LT, const float* __restrict__ RT,
                                       float* __restrict__ left, float* __restrict__ right) {
    __shared__ float se[DD];
    long long tok = blockIdx.x;
    int tid = threadIdx.x;
    const float* er = emb + tok * DD;
    for (int i = tid; i < DD; i += 256) se[i] = er[i];
    __syncthreads();
    int t = types[tok];
    const float* Lt = LT + (long long)t * DD * DD;
    const float* Rt = RT + (long long)t * DD * DD;
    for (int m = tid; m < DD; m += 256) {
        const float* lr = Lt + (long long)m * DD;
        const float* rr = Rt + (long long)m * DD;
        float al = 0.f, ar = 0.f;
        for (int d = 0; d < DD; d++) { float e = se[d]; al += e * lr[d]; ar += e * rr[d]; }
        left[tok * DD + m] = al;
        right[tok * DD + m] = ar;
    }
}

// ---------------- edge bias: ebias[b,n,m] = (LN_E(el[b,n]+er[b,m]) . Web + beb) / sqrt(2)
__global__ void ebias_kernel(const float* __restrict__ el, const float* __restrict__ er,
                             const float* __restrict__ lneg, const float* __restrict__ lneb,
                             const float* __restrict__ Web, const float* __restrict__ beb,
                             float* __restrict__ out) {
    long long i = (long long)blockIdx.x * 256 + threadIdx.x;
    if (i >= (long long)BB * SS * SS) return;
    int b = (int)(i / (SS * SS));
    long long rmd = i % (SS * SS);
    int nq = (int)(rmd / SS), mk = (int)(rmd % SS);
    const float* pl = el + ((long long)(b * SS) + nq) * EE;
    const float* pr = er + ((long long)(b * SS) + mk) * EE;
    float s = 0.f, q = 0.f;
    for (int e = 0; e < EE; e++) { float v = pl[e] + pr[e]; s += v; q += v * v; }
    float mean = s * (1.f / EE);
    float var = q * (1.f / EE) - mean * mean;
    float rstd = rsqrtf(var + 1e-5f);
    float acc = 0.f;
    for (int e = 0; e < EE; e++) {
        float v = pl[e] + pr[e];
        acc += ((v - mean) * rstd * lneg[e] + lneb[e]) * Web[e];
    }
    out[i] = (acc + beb[0]) * 0.70710678118654752f;
}

// ---------------- edge context: one block per (b,n) ----------------
// k_edge[m,e] = LN_E(el[b,n]+er[b,m]) @ Wke + bke   (built in LDS, bf16, [E][S])
// ectx[h,e]  = sum_m attn[b,h,n,m] * k_edge[m,e]    (WMMA: (16x S)@(S x 64), H=12 padded)
__global__ void ectx_kernel(const float* __restrict__ attn,
                            const float* __restrict__ el, const float* __restrict__ er,
                            const float* __restrict__ Wke, const float* __restrict__ bke,
                            const float* __restrict__ lneg, const float* __restrict__ lneb,
                            float* __restrict__ ectxr) {
    __shared__ __align__(16) u16 ke[EE * SS];   // 48 KB
    long long tok = blockIdx.x;
    int b = (int)(tok / SS), nq = (int)(tok % SS);
    int tid = threadIdx.x, wv = tid >> 5, lane = tid & 31;
    const float* elr = el + tok * EE;

    // phase 1: build k_edge in LDS, one wave per m
    for (int m = wv; m < SS; m += 8) {
        const float* err = er + ((long long)(b * SS) + m) * EE;
        float v0 = elr[lane] + err[lane];
        float v1 = elr[lane + 32] + err[lane + 32];
        float s = v0 + v1, q = v0 * v0 + v1 * v1;
        for (int o = 16; o > 0; o >>= 1) {
            s += __shfl_xor(s, o, 32);
            q += __shfl_xor(q, o, 32);
        }
        float mean = s * (1.f / EE);
        float var = q * (1.f / EE) - mean * mean;
        float rstd = rsqrtf(var + 1e-5f);
        float ne0 = (v0 - mean) * rstd * lneg[lane] + lneb[lane];
        float ne1 = (v1 - mean) * rstd * lneg[lane + 32] + lneb[lane + 32];
        float a0 = 0.f, a1 = 0.f;
        for (int j = 0; j < 32; j++) {
            float x0 = __shfl(ne0, j, 32);
            float x1 = __shfl(ne1, j, 32);
            a0 += x0 * Wke[j * EE + lane]        + x1 * Wke[(j + 32) * EE + lane];
            a1 += x0 * Wke[j * EE + lane + 32]   + x1 * Wke[(j + 32) * EE + lane + 32];
        }
        ke[lane * SS + m]        = f2bf(a0 + bke[lane]);
        ke[(lane + 32) * SS + m] = f2bf(a1 + bke[lane + 32]);
    }
    __syncthreads();

    // phase 2: waves 0..3 do WMMA; wave w owns e-tile [w*16, w*16+16)
    if (wv < 4) {
        int g = lane >> 4, r = lane & 15;
        int n0 = wv * 16;
        v8f acc = v8f{};
        for (int k = 0; k < SS; k += 32) {
            int kb0 = k + 8 * g;
            int kb1 = kb0 + 16;
            BFfrag a;
            if (r < HH) {
                const float* ar = attn + (((long long)(b * HH + r)) * SS + nq) * SS;
                float4 x0 = *(const float4*)(ar + kb0);
                float4 x1 = *(const float4*)(ar + kb0 + 4);
                float4 x2 = *(const float4*)(ar + kb1);
                float4 x3 = *(const float4*)(ar + kb1 + 4);
                pack8(a, 0, x0, x1);
                pack8(a, 8, x2, x3);
            } else {
                #pragma unroll
                for (int i = 0; i < 16; i++) a.u[i] = 0;
            }
            const u16* kr = &ke[(n0 + r) * SS];
            US8 u0 = *(const US8*)(kr + kb0);
            US8 u1 = *(const US8*)(kr + kb1);
            BFfrag bb;
            #pragma unroll
            for (int i = 0; i < 8; i++) { bb.u[i] = u0.s[i]; bb.u[8 + i] = u1.s[i]; }
            acc = __builtin_amdgcn_wmma_f32_16x16x32_bf16(
                false, a.b, false, bb.b, (short)0, acc, false, false);
        }
        #pragma unroll
        for (int v = 0; v < 8; v++) {
            int h = v + 8 * g;
            if (h < HH)
                ectxr[tok * (HH * EE) + h * EE + n0 + r] = acc[v];
        }
    }
}

// ---------------- classifier ----------------
__global__ void cls_kernel(const float* __restrict__ x, const float* __restrict__ w,
                           const float* __restrict__ b, float* __restrict__ out, int NCn) {
    __shared__ float red[256];
    int c = blockIdx.x % NCn;
    int bb = blockIdx.x / NCn;
    int tid = threadIdx.x;
    float a = 0.f;
    for (int d = tid; d < DD; d += 256)
        a += x[(long long)bb * SS * DD + d] * w[d * NCn + c];
    red[tid] = a; __syncthreads();
    for (int o = 128; o > 0; o >>= 1) { if (tid < o) red[tid] += red[tid + o]; __syncthreads(); }
    if (tid == 0) out[bb * NCn + c] = red[0] + b[c];
}

// ---------------- host orchestration ----------------
extern "C" void kernel_launch(void* const* d_in, const int* in_sizes, int n_in,
                              void* d_out, int out_size, void* d_ws, size_t ws_size,
                              hipStream_t stream) {
    const float* token_embs = (const float*)d_in[0];
    const int*   token_types = (const int*)d_in[1];
    const unsigned char* mask = (const unsigned char*)d_in[2];
    const float* LT   = (const float*)d_in[3];
    const float* RT   = (const float*)d_in[4];
    const float* edge_w = (const float*)d_in[5];
    const float* edge_b = (const float*)d_in[6];
    const float* Wq = (const float*)d_in[7];   const float* bq = (const float*)d_in[8];
    const float* Wk = (const float*)d_in[9];   const float* bk = (const float*)d_in[10];
    const float* Wv = (const float*)d_in[11];  const float* bv = (const float*)d_in[12];
    const float* Wke = (const float*)d_in[13]; const float* bke = (const float*)d_in[14];
    const float* Web = (const float*)d_in[15]; const float* beb = (const float*)d_in[16];
    const float* Weo = (const float*)d_in[17]; const float* beo = (const float*)d_in[18];
    const float* Wo = (const float*)d_in[19];  const float* bo = (const float*)d_in[20];
    const float* W1 = (const float*)d_in[21];  const float* b1 = (const float*)d_in[22];
    const float* W2 = (const float*)d_in[23];  const float* b2 = (const float*)d_in[24];
    const float* lnag = (const float*)d_in[25]; const float* lnab = (const float*)d_in[26];
    const float* lnfg = (const float*)d_in[27]; const float* lnfb = (const float*)d_in[28];
    const float* lneg = (const float*)d_in[29]; const float* lneb = (const float*)d_in[30];
    const float* cls_w = (const float*)d_in[31]; const float* cls_b = (const float*)d_in[32];
    float* out = (float*)d_out;

    // workspace bump allocator
    char* wsb = (char*)d_ws;
    size_t off = 0;
    auto alloc = [&](size_t bytes) -> void* {
        size_t o = (off + 255) & ~(size_t)255;
        off = o + bytes;
        return (void*)(wsb + o);
    };

    u16* ewT  = (u16*)alloc((size_t)EE * 2 * DD * 2);
    u16* wqT  = (u16*)alloc((size_t)DD * DD * 2);
    u16* wkT  = (u16*)alloc((size_t)DD * DD * 2);
    u16* wvT  = (u16*)alloc((size_t)DD * DD * 2);
    u16* weoT = (u16*)alloc((size_t)DD * DD * 2);
    u16* woT  = (u16*)alloc((size_t)DD * 2 * DD * 2);
    u16* w1T  = (u16*)alloc((size_t)FF * DD * 2);
    u16* w2T  = (u16*)alloc((size_t)DD * FF * 2);
    u16* kb   = (u16*)alloc((size_t)BSS * DD * 2);
    u16* vt   = (u16*)alloc((size_t)BSS * DD * 2);

    float* left  = (float*)alloc((size_t)BSS * DD * 4);
    float* right = (float*)alloc((size_t)BSS * DD * 4);
    float* el    = (float*)alloc((size_t)BSS * EE * 4);
    float* er    = (float*)alloc((size_t)BSS * EE * 4);
    float* nx    = (float*)alloc((size_t)BSS * DD * 4);
    float* q     = (float*)alloc((size_t)BSS * DD * 4);
    float* kk    = (float*)alloc((size_t)BSS * DD * 4);
    float* vv    = (float*)alloc((size_t)BSS * DD * 4);
    float* ctx   = (float*)alloc((size_t)BSS * DD * 4);
    float* ectxr = (float*)alloc((size_t)BSS * DD * 4);
    float* ectxp = (float*)alloc((size_t)BSS * DD * 4);
    float* hn    = (float*)alloc((size_t)BSS * DD * 4);
    float* xA    = (float*)alloc((size_t)BSS * DD * 4);
    float* xB    = (float*)alloc((size_t)BSS * DD * 4);
    float* gact  = (float*)alloc((size_t)BSS * FF * 4);
    float* scores = (float*)alloc((size_t)BB * HH * SS * SS * 4);
    float* ebias  = (float*)alloc((size_t)BB * SS * SS * 4);
    (void)ws_size; (void)in_sizes; (void)n_in; (void)out_size;

    auto cvtT = [&](const float* src, u16* dst, int K, int N) {
        long long tot = (long long)K * N;
        cvtT_kernel<<<dim3((unsigned)((tot + 255) / 256)), dim3(256), 0, stream>>>(src, dst, K, N);
    };
    auto gemm = [&](const float* A, long long az1, long long az2, int lda,
                    const u16* Bt, long long bz1, long long bz2, int ldb,
                    float* C, long long cz1, long long cz2, int ldc,
                    const float* add, long long dz1, long long dz2, int ldd,
                    const float* bias, float alpha, int gelu,
                    int M, int N, int K, int Z, int zdiv) {
        dim3 grid((unsigned)(N / 64), (unsigned)(M / 128), (unsigned)Z);
        wmma_gemm_kernel<<<grid, dim3(256), 0, stream>>>(
            A, az1, az2, lda, Bt, bz1, bz2, ldb, C, cz1, cz2, ldc,
            add, dz1, dz2, ldd, bias, alpha, gelu, K, zdiv);
    };

    const long long SD = (long long)SS * DD;       // 294912
    const long long SSq = (long long)SS * SS;      // 147456

    // ---- pre-layer ----
    cvtT(edge_w, ewT, 2 * DD, EE);
    token_transform_kernel<<<dim3(BSS), dim3(256), 0, stream>>>(
        token_embs, token_types, LT, RT, left, right);
    gemm(left, 0, 0, DD,  ewT, 0, 0, 2 * DD,  el, 0, 0, EE,
         nullptr, 0, 0, 0, edge_b, 1.f, 0, BSS, EE, DD, 1, 1);
    gemm(right, 0, 0, DD, ewT + DD, 0, 0, 2 * DD, er, 0, 0, EE,
         nullptr, 0, 0, 0, nullptr, 1.f, 0, BSS, EE, DD, 1, 1);

    const float* xcur = token_embs;

    for (int l = 0; l < LL; l++) {
        const long long DDl = (long long)DD * DD;
        cvtT(Wq + l * DDl, wqT, DD, DD);
        cvtT(Wk + l * DDl, wkT, DD, DD);
        cvtT(Wv + l * DDl, wvT, DD, DD);
        cvtT(Weo + l * DDl, weoT, DD, DD);
        cvtT(Wo + l * 2 * DDl, woT, 2 * DD, DD);
        cvtT(W1 + (long long)l * DD * FF, w1T, DD, FF);
        cvtT(W2 + (long long)l * FF * DD, w2T, FF, DD);

        // attention LN
        ln_rows_kernel<<<dim3(BSS), dim3(256), 0, stream>>>(xcur, lnag + l * DD, lnab + l * DD, nx, DD);

        // q,k,v
        gemm(nx, 0, 0, DD, wqT, 0, 0, DD, q, 0, 0, DD, nullptr, 0, 0, 0, bq + l * DD, 1.f, 0, BSS, DD, DD, 1, 1);
        gemm(nx, 0, 0, DD, wkT, 0, 0, DD, kk, 0, 0, DD, nullptr, 0, 0, 0, bk + l * DD, 1.f, 0, BSS, DD, DD, 1, 1);
        gemm(nx, 0, 0, DD, wvT, 0, 0, DD, vv, 0, 0, DD, nullptr, 0, 0, 0, bv + l * DD, 1.f, 0, BSS, DD, DD, 1, 1);

        long long nel = (long long)BSS * DD;
        cvtN_kernel<<<dim3((unsigned)((nel + 255) / 256)), dim3(256), 0, stream>>>(kk, kb, nel);
        transpose_v_kernel<<<dim3((unsigned)((nel + 255) / 256)), dim3(256), 0, stream>>>(vv, vt);

        // edge bias
        long long npair = (long long)BB * SS * SS;
        ebias_kernel<<<dim3((unsigned)((npair + 255) / 256)), dim3(256), 0, stream>>>(
            el, er, lneg + l * EE, lneb + l * EE, Web + l * EE, beb + l, ebias);

        // scores = q @ k^T * (2*DK)^-0.5 + ebias   (z = b*H + h)
        gemm(q, SD, DKK, DD,
             kb, SD, DKK, DD,
             scores, (long long)HH * SSq, SSq, SS,
             ebias, SSq, 0, SS,
             nullptr, 0.08838834764831845f, 0, SS, SS, DKK, BB * HH, HH);

        // softmax (masked), in-place
        softmax_kernel<<<dim3(BB * HH * SS), dim3(256), 0, stream>>>(scores, mask);

        // ctx = attn @ v  -> [B,S,D] (col h*DK+dk)
        gemm(scores, (long long)HH * SSq, SSq, SS,
             vt, (long long)HH * DKK * SS, (long long)DKK * SS, SS,
             ctx, SD, DKK, DD,
             nullptr, 0, 0, 0,
             nullptr, 1.f, 0, SS, DKK, SS, BB * HH, HH);

        // edge context -> ectxr [B,S,H*E]
        ectx_kernel<<<dim3(BSS), dim3(256), 0, stream>>>(
            scores, el, er, Wke + (long long)l * EE * EE, bke + l * EE,
            lneg + l * EE, lneb + l * EE, ectxr);

        // ectxp = ectxr @ Weo + beo
        gemm(ectxr, 0, 0, DD, weoT, 0, 0, DD, ectxp, 0, 0, DD,
             nullptr, 0, 0, 0, beo + l * DD, 1.f, 0, BSS, DD, DD, 1, 1);

        // x_mid = xcur + ctx @ Wo_top + bo ; x_mid += ectxp @ Wo_bot
        float* xmid = xA;
        gemm(ctx, 0, 0, DD, woT, 0, 0, 2 * DD, xmid, 0, 0, DD,
             xcur, 0, 0, DD, bo + l * DD, 1.f, 0, BSS, DD, DD, 1, 1);
        gemm(ectxp, 0, 0, DD, woT + DD, 0, 0, 2 * DD, xmid, 0, 0, DD,
             xmid, 0, 0, DD, nullptr, 1.f, 0, BSS, DD, DD, 1, 1);

        // FFN
        ln_rows_kernel<<<dim3(BSS), dim3(256), 0, stream>>>(xmid, lnfg + l * DD, lnfb + l * DD, hn, DD);
        gemm(hn, 0, 0, DD, w1T, 0, 0, DD, gact, 0, 0, FF,
             nullptr, 0, 0, 0, b1 + l * FF, 1.f, 1 /*gelu*/, BSS, FF, DD, 1, 1);
        float* xnew = xB;
        gemm(gact, 0, 0, FF, w2T, 0, 0, FF, xnew, 0, 0, DD,
             xmid, 0, 0, DD, b2 + l * DD, 1.f, 0, BSS, DD, FF, 1, 1);

        xcur = xnew;
    }

    // classifier: out[b,c] = x[b,0,:] . cls_w[:,c] + cls_b[c]
    cls_kernel<<<dim3(BB * 2), dim3(256), 0, stream>>>(xcur, cls_w, cls_b, out, 2);
}
